// SimplePointNetPP_40561671143746
// MI455X (gfx1250) — compile-verified, matrix operand order checked by reading, not run on hardware
//
#include <hip/hip_runtime.h>

// ---------------------------------------------------------------------------
// PointNet++ (2-level set abstraction) for MI455X / gfx1250, wave32 + WMMA.
// B=8 clouds, N=2048 pts, K=64 neighbors.
// GEMMs run as v_wmma_f32_16x16x32_bf16 with BN folded into layer-1 weights.
// Per-wave tile loops use readfirstlane so loop control is scalar (SGPR) and
// EXEC is structurally all-ones around every WMMA.
// ---------------------------------------------------------------------------

typedef __attribute__((ext_vector_type(16))) __bf16 v16bf;
typedef __attribute__((ext_vector_type(8)))  float  v8f;

#define BB 8
#define NN 2048
#define KNBR 64

// ---- WMMA fragment helpers -------------------------------------------------
// A-matrix 16x32 bf16 layout (ISA 7.12.2): lane L (row = L&15, half = L>>4),
// VGPR v<4 -> K = 2v + 8*half ; v>=4 -> K = 16 + 2(v-4) + 8*half (pairs).
__device__ __forceinline__ v16bf load_A_frag(const __bf16* base, int ldk, int lane) {
  int row = lane & 15, hf = lane >> 4;
  const __bf16* pr = base + row * ldk + hf * 8;
  union { unsigned int u[8]; v16bf h; } c;
  const unsigned int* p0 = (const unsigned int*)pr;        // k pairs 0..7 (+8*half)
  const unsigned int* p1 = (const unsigned int*)(pr + 16); // k pairs 16..23 (+8*half)
  c.u[0] = p0[0]; c.u[1] = p0[1]; c.u[2] = p0[2]; c.u[3] = p0[3];
  c.u[4] = p1[0]; c.u[5] = p1[1]; c.u[6] = p1[2]; c.u[7] = p1[3];
  return c.h;
}

// B fragments are pre-swizzled to memory so each lane loads 16 contiguous bf16.
__device__ __forceinline__ v16bf load_B_frag(const __bf16* p) {
  return *(const v16bf*)p;
}

// ---- normalize_pos: per-cloud mean / unbiased std --------------------------
__global__ __launch_bounds__(256) void normalize_kernel(const float* __restrict__ pos,
                                                        float* __restrict__ out) {
  __shared__ float red[768];
  __shared__ float smean[3], sinv[3];
  int b = blockIdx.x, tid = threadIdx.x;
  const float* p = pos + (size_t)b * NN * 3;
  float s0 = 0.f, s1 = 0.f, s2 = 0.f;
  for (int j = tid; j < NN; j += 256) { s0 += p[j*3]; s1 += p[j*3+1]; s2 += p[j*3+2]; }
  red[tid] = s0; red[256+tid] = s1; red[512+tid] = s2; __syncthreads();
  for (int s = 128; s > 0; s >>= 1) {
    if (tid < s) { red[tid]+=red[tid+s]; red[256+tid]+=red[256+tid+s]; red[512+tid]+=red[512+tid+s]; }
    __syncthreads();
  }
  if (tid == 0) { smean[0]=red[0]/NN; smean[1]=red[256]/NN; smean[2]=red[512]/NN; }
  __syncthreads();
  float m0 = smean[0], m1 = smean[1], m2 = smean[2];
  s0 = s1 = s2 = 0.f;
  for (int j = tid; j < NN; j += 256) {
    float d0=p[j*3]-m0, d1=p[j*3+1]-m1, d2=p[j*3+2]-m2;
    s0 += d0*d0; s1 += d1*d1; s2 += d2*d2;
  }
  red[tid]=s0; red[256+tid]=s1; red[512+tid]=s2; __syncthreads();
  for (int s = 128; s > 0; s >>= 1) {
    if (tid < s) { red[tid]+=red[tid+s]; red[256+tid]+=red[256+tid+s]; red[512+tid]+=red[512+tid+s]; }
    __syncthreads();
  }
  if (tid == 0) {
    sinv[0] = 1.f / (sqrtf(red[0]  /(NN-1)) + 1e-6f);
    sinv[1] = 1.f / (sqrtf(red[256]/(NN-1)) + 1e-6f);
    sinv[2] = 1.f / (sqrtf(red[512]/(NN-1)) + 1e-6f);
  }
  __syncthreads();
  float i0 = sinv[0], i1 = sinv[1], i2 = sinv[2];
  float* o = out + (size_t)b * NN * 3;
  for (int j = tid; j < NN; j += 256) {
    o[j*3]   = (p[j*3]  -m0)*i0;
    o[j*3+1] = (p[j*3+1]-m1)*i1;
    o[j*3+2] = (p[j*3+2]-m2)*i2;
  }
}

// ---- Farthest point sampling ----------------------------------------------
template<int NPTS, int MSEL>
__global__ __launch_bounds__(256) void fps_kernel(const float* __restrict__ pos,
                                                  int* __restrict__ out) {
  constexpr int PPT = NPTS / 256;
  __shared__ float wbd[8]; __shared__ int wbi[8];
  __shared__ float lastp[3];
  int b = blockIdx.x, tid = threadIdx.x, lane = tid & 31;
  int wv = __builtin_amdgcn_readfirstlane(tid >> 5);
  const float* p = pos + (size_t)b * NPTS * 3;
  float dist[PPT];
#pragma unroll
  for (int t = 0; t < PPT; ++t) dist[t] = 1e10f;
  if (tid == 0) { out[(size_t)b*MSEL] = 0; lastp[0]=p[0]; lastp[1]=p[1]; lastp[2]=p[2]; }
  __syncthreads();
  for (int k = 1; k < MSEL; ++k) {
    float lx = lastp[0], ly = lastp[1], lz = lastp[2];
    float bk = -1.f; int bi = 0;
#pragma unroll
    for (int t = 0; t < PPT; ++t) {
      int j = tid + t * 256;
      float dx = p[j*3]-lx, dy = p[j*3+1]-ly, dz = p[j*3+2]-lz;
      float d = dx*dx + dy*dy + dz*dz;
      dist[t] = fminf(dist[t], d);
      if (dist[t] > bk) { bk = dist[t]; bi = j; }
    }
    for (int off = 16; off; off >>= 1) {
      float ok = __shfl_xor(bk, off, 32); int oi = __shfl_xor(bi, off, 32);
      if (ok > bk || (ok == bk && oi < bi)) { bk = ok; bi = oi; }
    }
    if (lane == 0) { wbd[wv] = bk; wbi[wv] = bi; }
    __syncthreads();
    if (tid == 0) {
      float best = wbd[0]; int besti = wbi[0];
      for (int w = 1; w < 8; ++w) if (wbd[w] > best) { best = wbd[w]; besti = wbi[w]; }
      out[(size_t)b*MSEL + k] = besti;
      lastp[0]=p[besti*3]; lastp[1]=p[besti*3+1]; lastp[2]=p[besti*3+2];
    }
    __syncthreads();
  }
}

// ---- gathers ---------------------------------------------------------------
__global__ void gather_pos_kernel(const float* __restrict__ src, const int* __restrict__ idx,
                                  float* __restrict__ dst, int Msrc, int Mdst) {
  int i = blockIdx.x * blockDim.x + threadIdx.x;
  if (i >= BB * Mdst) return;
  int b = i / Mdst;
  int j = idx[i];
  const float* s = src + ((size_t)b * Msrc + j) * 3;
  dst[i*3] = s[0]; dst[i*3+1] = s[1]; dst[i*3+2] = s[2];
}

__global__ void gather_feat_kernel(const __bf16* __restrict__ src, const int* __restrict__ idx,
                                   __bf16* __restrict__ dst, int Msrc, int Mdst, int C) {
  int i = blockIdx.x * blockDim.x + threadIdx.x;
  if (i >= BB * Mdst * C) return;
  int c = i % C, t = i / C;
  int b = t / Mdst;
  int j = idx[t];
  dst[i] = src[((size_t)b * Msrc + j) * C + c];
}

// ---- radius ball query: K nearest within r (self included) -----------------
template<int M>
__global__ __launch_bounds__(256) void ball_kernel(const float* __restrict__ pos,
                                                   int* __restrict__ nbr, float r2) {
  constexpr int CPL = M / 32;
  int lane = threadIdx.x & 31;
  int wv = __builtin_amdgcn_readfirstlane(threadIdx.x >> 5);
  int center = blockIdx.x * 8 + wv;                 // scalar
  int b = center / M, ci = center - b * M;          // scalar
  const float* p = pos + (size_t)b * M * 3;
  float cx = p[ci*3], cy = p[ci*3+1], cz = p[ci*3+2];
  float d2[CPL];
#pragma unroll
  for (int t = 0; t < CPL; ++t) {
    int j = lane + 32 * t;
    float dx = p[j*3]-cx, dy = p[j*3+1]-cy, dz = p[j*3+2]-cz;
    d2[t] = dx*dx + dy*dy + dz*dz;
  }
  int* out = nbr + (size_t)center * KNBR;
  for (int k = 0; k < KNBR; ++k) {
    float bk = 3e38f; int bi = -1;
#pragma unroll
    for (int t = 0; t < CPL; ++t) {
      int j = lane + 32 * t;
      if (d2[t] < bk) { bk = d2[t]; bi = j; }
    }
    for (int off = 16; off; off >>= 1) {
      float ok = __shfl_xor(bk, off, 32); int oi = __shfl_xor(bi, off, 32);
      if (ok < bk || (ok == bk && oi < bi)) { bk = ok; bi = oi; }
    }
    bool good = (bi >= 0) && (bk <= r2);
    if (good) {
      if (lane == 0) out[k] = bi;
#pragma unroll
      for (int t = 0; t < CPL; ++t)
        if (lane + 32 * t == bi) d2[t] = 3e38f;
    } else {
      for (int kk = k + lane; kk < KNBR; kk += 32) out[kk] = -1;
      break;
    }
  }
}

// ---- weight prep: BN-fold + bf16 + fragment swizzle ------------------------
// Output layout: tile (nt,kt) at ((nt*KT + kt)*512), within tile lane*16 + 2v+e,
// matching load_B_frag's per-lane contiguous 32-byte read.
__global__ void swizzle_w_kernel(const float* __restrict__ W, const float* __restrict__ g,
                                 const float* __restrict__ vvar, __bf16* __restrict__ out,
                                 int Kreal, int KP, int Nc, int useScale) {
  int KT = KP / 32;
  int total = (KP / 32) * (Nc / 16) * 512;
  for (int idx = blockIdx.x * blockDim.x + threadIdx.x; idx < total;
       idx += gridDim.x * blockDim.x) {
    int t = idx >> 9, r = idx & 511;
    int nt = t / KT, kt = t - nt * KT;
    int lane = r >> 4, j = r & 15;
    int v = j >> 1, e = j & 1;
    int hf = lane >> 4;
    int kb = (v < 4 ? 2*v : 16 + 2*(v-4)) + 8*hf;
    int k = kt*32 + kb + e;
    int n = nt*16 + (lane & 15);
    float val = 0.f;
    if (k < Kreal) {
      val = W[(size_t)k * Nc + n];
      if (useScale) val *= g[n] * rsqrtf(vvar[n] + 1e-5f);
    }
    out[idx] = (__bf16)val;
  }
}

__global__ void fold_bias_kernel(const float* b1, const float* g, const float* beta,
                                 const float* m, const float* vvar, float* out, int n) {
  int i = blockIdx.x * blockDim.x + threadIdx.x;
  if (i < n) out[i] = (b1[i] - m[i]) * g[i] * rsqrtf(vvar[i] + 1e-5f) + beta[i];
}

// ---- fused PointNetConv: gather -> GEMM1(+BN+ReLU) -> GEMM2(+ReLU) -> max --
template<int CX, int KP1, int C1, int C2, bool XF32, bool FINAL>
__global__ __launch_bounds__(256) void conv_kernel(
    const float* __restrict__ pos, const float* __restrict__ xinf,
    const __bf16* __restrict__ xinb, const int* __restrict__ nbr,
    const __bf16* __restrict__ w1s, const float* __restrict__ b1,
    const __bf16* __restrict__ w2s, const float* __restrict__ b2,
    __bf16* __restrict__ outb, float* __restrict__ outf, int M) {
  constexpr int KT1 = KP1 / 32, NT1 = C1 / 16, KT2 = C1 / 32, NT2 = C2 / 16;
  __shared__ __bf16 sFeat[64 * KP1];
  __shared__ __bf16 sH[64 * C1];
  __shared__ int sNbr[64];
  __shared__ float sCtr[3];
  int center = blockIdx.x;
  int b = center / M, ci = center - b * M;
  int tid = threadIdx.x;
  const float* posB = pos + (size_t)b * M * 3;
  if (tid < 64) sNbr[tid] = nbr[(size_t)center * KNBR + tid];
  if (tid == 64) { sCtr[0]=posB[ci*3]; sCtr[1]=posB[ci*3+1]; sCtr[2]=posB[ci*3+2]; }
  __syncthreads();
  // gather neighbor features (zeros for pad/invalid rows; masked later)
  for (int i = tid; i < 64 * KP1; i += 256) {
    int r = i / KP1, c = i - r * KP1;
    int j = sNbr[r];
    float v = 0.f;
    if (j >= 0) {
      if (c < CX) {
        if constexpr (XF32) v = xinf[((size_t)b * M + j) * CX + c];
        else                v = (float)xinb[((size_t)b * M + j) * CX + c];
      } else if (c < CX + 3) {
        v = posB[j*3 + (c - CX)] - sCtr[c - CX];
      }
    }
    sFeat[i] = (__bf16)v;
  }
  __syncthreads();
  int lane = tid & 31;
  int wv = __builtin_amdgcn_readfirstlane(tid >> 5);  // scalar wave id
  // ---- stage 1: [64 x KP1] @ [KP1 x C1], bias+ReLU -> bf16 LDS
  for (int t = wv; t < 4 * NT1; t += 8) {             // scalar loop control
    int mt = t & 3, nt = t >> 2;
    v8f acc = {};
    for (int kt = 0; kt < KT1; ++kt) {
      v16bf a  = load_A_frag(sFeat + (mt * 16) * KP1 + kt * 32, KP1, lane);
      v16bf bf = load_B_frag(w1s + ((size_t)(nt * KT1 + kt)) * 512 + lane * 16);
      acc = __builtin_amdgcn_wmma_f32_16x16x32_bf16(false, a, false, bf, (short)0, acc,
                                                    false, false);
    }
    int col = nt * 16 + (lane & 15);
    int hf = lane >> 4;
    float bias = b1[col];
#pragma unroll
    for (int r = 0; r < 8; ++r) {
      float h = fmaxf(acc[r] + bias, 0.f);
      int row = mt * 16 + r + 8 * hf;  // C/D layout: VGPR r -> M = r + 8*half
      sH[row * C1 + col] = (__bf16)h;
    }
  }
  __syncthreads();
  // ---- stage 2: [64 x C1] @ [C1 x C2], ReLU, masked column-max over K
  for (int nt = wv; nt < NT2; nt += 8) {              // scalar loop control
    float cmax = -INFINITY;
    int col = nt * 16 + (lane & 15);
    int hf = lane >> 4;
    float bias = b2[col];
    for (int mt = 0; mt < 4; ++mt) {
      v8f acc = {};
      for (int kt = 0; kt < KT2; ++kt) {
        v16bf a  = load_A_frag(sH + (mt * 16) * C1 + kt * 32, C1, lane);
        v16bf bf = load_B_frag(w2s + ((size_t)(nt * KT2 + kt)) * 512 + lane * 16);
        acc = __builtin_amdgcn_wmma_f32_16x16x32_bf16(false, a, false, bf, (short)0, acc,
                                                      false, false);
      }
#pragma unroll
      for (int r = 0; r < 8; ++r) {
        int row = mt * 16 + r + 8 * hf;
        float h = fmaxf(acc[r] + bias, 0.f);
        if (sNbr[row] < 0) h = -INFINITY;
        cmax = fmaxf(cmax, h);
      }
    }
    cmax = fmaxf(cmax, __shfl_xor(cmax, 16, 32));  // merge the two row-halves
    if (lane < 16) {
      if constexpr (FINAL) outf[(size_t)center * C2 + col] = cmax;
      else                 outb[(size_t)center * C2 + col] = (__bf16)cmax;
    }
  }
}

// ---- batch2 tail -----------------------------------------------------------
__global__ void batch_fill_kernel(int* out) {
  int i = blockIdx.x * blockDim.x + threadIdx.x;
  if (i < BB * (NN / 4)) out[i] = i >> 9;  // i / 512
}

// ---------------------------------------------------------------------------
extern "C" void kernel_launch(void* const* d_in, const int* in_sizes, int n_in,
                              void* d_out, int out_size, void* d_ws, size_t ws_size,
                              hipStream_t stream) {
  (void)in_sizes; (void)n_in; (void)out_size; (void)ws_size;
  const float* pos   = (const float*)d_in[0];
  const float* c1_w1 = (const float*)d_in[1];
  const float* c1_b1 = (const float*)d_in[2];
  const float* c1_g  = (const float*)d_in[3];
  const float* c1_be = (const float*)d_in[4];
  const float* c1_m  = (const float*)d_in[5];
  const float* c1_v  = (const float*)d_in[6];
  const float* c1_w2 = (const float*)d_in[7];
  const float* c1_b2 = (const float*)d_in[8];
  const float* c2_w1 = (const float*)d_in[9];
  const float* c2_b1 = (const float*)d_in[10];
  const float* c2_g  = (const float*)d_in[11];
  const float* c2_be = (const float*)d_in[12];
  const float* c2_m  = (const float*)d_in[13];
  const float* c2_v  = (const float*)d_in[14];
  const float* c2_w2 = (const float*)d_in[15];
  const float* c2_b2 = (const float*)d_in[16];

  char* ws = (char*)d_ws;
  size_t off = 0;
  auto alloc = [&](size_t bytes) -> char* {
    char* r = ws + off;
    off += (bytes + 255) & ~(size_t)255;
    return r;
  };
  float*  posn  = (float*)alloc((size_t)BB * NN * 3 * 4);
  int*    idx1  = (int*)  alloc((size_t)BB * (NN/2) * 4);
  float*  pos1  = (float*)alloc((size_t)BB * (NN/2) * 3 * 4);
  int*    nbr1  = (int*)  alloc((size_t)BB * (NN/2) * KNBR * 4);
  __bf16* x1    = (__bf16*)alloc((size_t)BB * (NN/2) * 128 * 2);
  int*    idx2  = (int*)  alloc((size_t)BB * (NN/4) * 4);
  float*  pos2  = (float*)alloc((size_t)BB * (NN/4) * 3 * 4);
  __bf16* x2in  = (__bf16*)alloc((size_t)BB * (NN/4) * 128 * 2);
  int*    nbr2  = (int*)  alloc((size_t)BB * (NN/4) * KNBR * 4);
  __bf16* w1s1  = (__bf16*)alloc((size_t)(32/32)  * (128/16) * 512 * 2);
  __bf16* w2s1  = (__bf16*)alloc((size_t)(128/32) * (128/16) * 512 * 2);
  __bf16* w1s2  = (__bf16*)alloc((size_t)(160/32) * (256/16) * 512 * 2);
  __bf16* w2s2  = (__bf16*)alloc((size_t)(256/32) * (512/16) * 512 * 2);
  float*  b1f1  = (float*)alloc(128 * 4);
  float*  b1f2  = (float*)alloc(256 * 4);

  // 0) normalize positions per cloud
  normalize_kernel<<<BB, 256, 0, stream>>>(pos, posn);
  // 1) weight prep (independent of geometry)
  fold_bias_kernel<<<1, 128, 0, stream>>>(c1_b1, c1_g, c1_be, c1_m, c1_v, b1f1, 128);
  fold_bias_kernel<<<1, 256, 0, stream>>>(c2_b1, c2_g, c2_be, c2_m, c2_v, b1f2, 256);
  swizzle_w_kernel<<<64, 256, 0, stream>>>(c1_w1, c1_g, c1_v, w1s1, 6, 32, 128, 1);
  swizzle_w_kernel<<<64, 256, 0, stream>>>(c1_w2, nullptr, nullptr, w2s1, 128, 128, 128, 0);
  swizzle_w_kernel<<<64, 256, 0, stream>>>(c2_w1, c2_g, c2_v, w1s2, 131, 160, 256, 1);
  swizzle_w_kernel<<<256, 256, 0, stream>>>(c2_w2, nullptr, nullptr, w2s2, 256, 256, 512, 0);
  // 2) level 1: FPS -> gather -> ball query (r=0.1) -> conv (6->128->128)
  fps_kernel<NN, NN/2><<<BB, 256, 0, stream>>>(posn, idx1);
  gather_pos_kernel<<<(BB*(NN/2) + 255)/256, 256, 0, stream>>>(posn, idx1, pos1, NN, NN/2);
  ball_kernel<NN/2><<<BB*(NN/2)/8, 256, 0, stream>>>(pos1, nbr1, 0.1f * 0.1f);
  conv_kernel<3, 32, 128, 128, true, false><<<BB*(NN/2), 256, 0, stream>>>(
      pos1, pos1, nullptr, nbr1, w1s1, b1f1, w2s1, c1_b2, x1, nullptr, NN/2);
  // 3) level 2: FPS -> gather pos+feat -> ball query (r=0.25) -> conv (131->256->512)
  fps_kernel<NN/2, NN/4><<<BB, 256, 0, stream>>>(pos1, idx2);
  gather_pos_kernel<<<(BB*(NN/4) + 255)/256, 256, 0, stream>>>(pos1, idx2, pos2, NN/2, NN/4);
  gather_feat_kernel<<<(BB*(NN/4)*128 + 255)/256, 256, 0, stream>>>(x1, idx2, x2in,
                                                                    NN/2, NN/4, 128);
  ball_kernel<NN/4><<<BB*(NN/4)/8, 256, 0, stream>>>(pos2, nbr2, 0.25f * 0.25f);
  float* outx2 = (float*)d_out;
  conv_kernel<128, 160, 256, 512, false, true><<<BB*(NN/4), 256, 0, stream>>>(
      pos2, nullptr, x2in, nbr2, w1s2, b1f2, w2s2, c2_b2, nullptr, outx2, NN/4);
  // 4) batch2 tail (int32 ids packed after the 8*512*512 floats)
  int* bt = (int*)d_out + (size_t)BB * (NN/4) * 512;
  batch_fill_kernel<<<(BB*(NN/4) + 255)/256, 256, 0, stream>>>(bt);
}
